// VRWKV_SpatialMix_62921270886819
// MI455X (gfx1250) — matrix operand order repaired and checked by reading, not compiled, not run on hardware
//
#include <hip/hip_runtime.h>
#include <stdint.h>

typedef __attribute__((ext_vector_type(16))) __bf16 v16bf;
typedef __attribute__((ext_vector_type(8)))  float  v8f;

__device__ __forceinline__ unsigned short f2bf(float f) {
    unsigned int u = __float_as_uint(f);
    u += 0x7FFFu + ((u >> 16) & 1u);   // round-to-nearest-even
    return (unsigned short)(u >> 16);
}

// gfx1250 async global->LDS copy, 16B per lane (ASYNCcnt-tracked)
__device__ __forceinline__ void async_copy16(const void* gaddr, unsigned lds_off) {
    asm volatile("global_load_async_to_lds_b128 %0, %1, off"
                 :: "v"(lds_off), "v"(gaddr)
                 : "memory");
}
__device__ __forceinline__ void wait_async0() {
    asm volatile("s_wait_asynccnt 0x0" ::: "memory");
}

// ---------------------------------------------------------------------------
// 1) Convert the four weight matrices f32 -> bf16 (layout [N,K], K contiguous)
// ---------------------------------------------------------------------------
__global__ void convert_weights_kernel(const float* __restrict__ wk,
                                       const float* __restrict__ wv,
                                       const float* __restrict__ wr,
                                       const float* __restrict__ wo,
                                       unsigned short* __restrict__ ok,
                                       unsigned short* __restrict__ ov,
                                       unsigned short* __restrict__ orr,
                                       unsigned short* __restrict__ oo,
                                       int n) {
    int idx = blockIdx.x * blockDim.x + threadIdx.x;
    if (idx >= 4 * n) return;
    int w = idx / n, j = idx - w * n;
    const float*    s = (w == 0) ? wk : (w == 1) ? wv : (w == 2) ? wr : wo;
    unsigned short* d = (w == 0) ? ok : (w == 1) ? ov : (w == 2) ? orr : oo;
    d[j] = f2bf(s[j]);
}

// ---------------------------------------------------------------------------
// 2) token_shift + channel mixing, output bf16 activations xk/xv/xr
// ---------------------------------------------------------------------------
__global__ void mix_shift_kernel(const float* __restrict__ x,
                                 const float* __restrict__ mk,
                                 const float* __restrict__ mv,
                                 const float* __restrict__ mr,
                                 unsigned short* __restrict__ xk,
                                 unsigned short* __restrict__ xv,
                                 unsigned short* __restrict__ xr,
                                 int T, int C, long long N) {
    long long idx = (long long)blockIdx.x * blockDim.x + threadIdx.x;
    if (idx >= N) return;
    int c = (int)(idx % C);
    long long bt = idx / C;
    int t = (int)(bt % T);
    float xc = x[idx];
    float xx;
    int g = C >> 2;                       // C_GAMMA = 0.25
    if (c < g)            xx = (t > 0)     ? x[idx - C] : 0.f;   // shift fwd
    else if (c < (g<<1))  xx = (t + 1 < T) ? x[idx + C] : 0.f;   // shift bwd
    else                  xx = xc;                                // passthrough
    float a;
    a = mk[c]; xk[idx] = f2bf(xc * a + xx * (1.f - a));
    a = mv[c]; xv[idx] = f2bf(xc * a + xx * (1.f - a));
    a = mr[c]; xr[idx] = f2bf(xc * a + xx * (1.f - a));
}

// ---------------------------------------------------------------------------
// 3) bf16 WMMA GEMM:  out[M,N] = A[M,K] * W[N,K]^T   (f32 accumulate)
//    Block tile 128x128, K chunk 32, 256 threads (8 waves),
//    wave tile 64x32 => 4x2 v_wmma_f32_16x16x32_bf16 accumulators.
//    Double-buffered LDS staging via async global->LDS (ASYNCcnt).
// ---------------------------------------------------------------------------
#define LDS_STRIDE 40   // 32 bf16 + pad
#define BUF_USHORT (128 * LDS_STRIDE)

__global__ __launch_bounds__(256) void gemm_bf16_wmma_kernel(
        const unsigned short* __restrict__ A,   // [M,K] bf16 row-major
        const unsigned short* __restrict__ W,   // [N,K] bf16 row-major
        float* __restrict__ out,                // [M,N] f32
        int M, int N, int K, int epi) {
    __shared__ unsigned short As[2 * BUF_USHORT];
    __shared__ unsigned short Bs[2 * BUF_USHORT];

    const int tid  = threadIdx.x;
    const int wave = tid >> 5;
    const int lane = tid & 31;
    const int half = lane >> 4;    // 0: lanes 0-15, 1: lanes 16-31
    const int l16  = lane & 15;
    const int wm   = wave & 1;     // 2 row groups * 64 rows
    const int wn   = wave >> 1;    // 4 col groups * 32 cols

    const long long bm = (long long)blockIdx.x * 128;
    const int       bn = blockIdx.y * 128;

    const int srow = tid >> 1;     // staging row 0..127
    const int shalf = tid & 1;     // staging 16-bf16 half of a 32-col row

    // per-thread staging source/dest (advance kc by +32 each chunk)
    const unsigned short* gA = A + (bm + srow) * (long long)K + shalf * 16;
    const unsigned short* gB = W + (long long)(bn + srow) * K + shalf * 16;
    const unsigned lA = (unsigned)(unsigned long long)(const void*)
                        (As + srow * LDS_STRIDE + shalf * 16);
    const unsigned lB = (unsigned)(unsigned long long)(const void*)
                        (Bs + srow * LDS_STRIDE + shalf * 16);

    v8f acc[4][2];
    #pragma unroll
    for (int i = 0; i < 4; ++i)
        #pragma unroll
        for (int j = 0; j < 2; ++j)
            acc[i][j] = (v8f){0.f,0.f,0.f,0.f,0.f,0.f,0.f,0.f};

    const int NC = K >> 5;         // K chunks of 32

    // prologue: stage chunk 0 into buffer 0
    async_copy16(gA, lA);
    async_copy16(gA + 8, lA + 16);
    async_copy16(gB, lB);
    async_copy16(gB + 8, lB + 16);

    for (int ic = 0; ic < NC; ++ic) {
        wait_async0();             // our chunk-ic fills have landed in LDS
        __syncthreads();           // everyone's fills visible; prev reads done

        // stage chunk ic+1 into the other buffer (overlaps with compute)
        if (ic + 1 < NC) {
            int koff = (ic + 1) << 5;
            unsigned dsel = ((ic + 1) & 1) * (BUF_USHORT * 2);   // byte offset
            async_copy16(gA + koff, lA + dsel);
            async_copy16(gA + koff + 8, lA + dsel + 16);
            async_copy16(gB + koff, lB + dsel);
            async_copy16(gB + koff + 8, lB + dsel + 16);
        }

        const unsigned short* curA = As + (ic & 1) * BUF_USHORT;
        const unsigned short* curB = Bs + (ic & 1) * BUF_USHORT;

        // ---- build fragments in contiguous k order (k = 16*half + s), same for A and B
        union AF { v16bf v; unsigned int u[8]; } af[4];
        union BF { v16bf v; uint4 q[2]; }       bf2[2];

        #pragma unroll
        for (int mt = 0; mt < 4; ++mt) {
            int m = wm * 64 + mt * 16 + l16;
            const unsigned short* ap = curA + m * LDS_STRIDE;
            #pragma unroll
            for (int vv = 0; vv < 8; ++vv) {
                // 16-bit A 16x32: VGPR vv holds K = {base, base+1}
                int k0 = (half << 4) + (vv << 1);
                af[mt].u[vv] = *(const unsigned int*)(ap + k0);
            }
        }
        #pragma unroll
        for (int nt = 0; nt < 2; ++nt) {
            int n = wn * 32 + nt * 16 + l16;
            // 16-bit B 32x16: lane half selects K 0-15 / 16-31, contiguous
            const uint4* bp = (const uint4*)(curB + n * LDS_STRIDE + half * 16);
            bf2[nt].q[0] = bp[0];
            bf2[nt].q[1] = bp[1];
        }

        #pragma unroll
        for (int mt = 0; mt < 4; ++mt)
            #pragma unroll
            for (int nt = 0; nt < 2; ++nt)
                acc[mt][nt] = __builtin_amdgcn_wmma_f32_16x16x32_bf16(
                    false, af[mt].v, false, bf2[nt].v,
                    (short)0, acc[mt][nt], false, false);
    }

    // ---- epilogue: C/D layout: VGPR r -> row (r + 8*half), col = lane&15
    #pragma unroll
    for (int mt = 0; mt < 4; ++mt)
        #pragma unroll
        for (int nt = 0; nt < 2; ++nt) {
            int ncol = bn + wn * 32 + nt * 16 + l16;
            #pragma unroll
            for (int r = 0; r < 8; ++r) {
                long long mrow = bm + wm * 64 + mt * 16 + r + half * 8;
                float val = acc[mt][nt][r];
                if (epi == 1) val = 1.f / (1.f + __expf(-val));
                out[mrow * N + ncol] = val;
            }
        }
}

// ---------------------------------------------------------------------------
// 4a) forward bi-WKV scan: store strict-prefix state (a, b, p) per token
// ---------------------------------------------------------------------------
__global__ void scan_fwd_kernel(const float* __restrict__ k,
                                const float* __restrict__ v,
                                const float* __restrict__ decay,
                                float* __restrict__ af,
                                float* __restrict__ bfv,
                                float* __restrict__ pf,
                                int B, int T, int C) {
    int idx = blockIdx.x * blockDim.x + threadIdx.x;
    if (idx >= B * C) return;
    int b = idx / C, c = idx - b * C;
    float w = decay[c] / (float)T;
    float a = 0.f, bb = 0.f, p = -1e38f;
    size_t base = (size_t)b * T * C + c;
    for (int t = 0; t < T; ++t) {
        size_t off = base + (size_t)t * C;
        af[off] = a; bfv[off] = bb; pf[off] = p;       // state excludes token t
        float kt = k[off], vt = v[off];
        float m  = fmaxf(p - w, kt);
        float e1 = __expf(p - w - m);
        float e2 = __expf(kt - m);
        a  = a * e1 + vt * e2;
        bb = bb * e1 + e2;
        p  = m;
    }
}

// ---------------------------------------------------------------------------
// 4b) backward scan + combine + sigmoid gate; emit bf16 (sr * rwkv)
// ---------------------------------------------------------------------------
__global__ void scan_bwd_combine_kernel(const float* __restrict__ k,
                                        const float* __restrict__ v,
                                        const float* __restrict__ decay,
                                        const float* __restrict__ first,
                                        const float* __restrict__ af,
                                        const float* __restrict__ bfv,
                                        const float* __restrict__ pf,
                                        const float* __restrict__ sr,
                                        unsigned short* __restrict__ outb,
                                        int B, int T, int C) {
    int idx = blockIdx.x * blockDim.x + threadIdx.x;
    if (idx >= B * C) return;
    int b = idx / C, c = idx - b * C;
    float w = decay[c] / (float)T;
    float u = first[c] / (float)T;
    float a = 0.f, bb = 0.f, p = -1e38f;               // strict-suffix state
    size_t base = (size_t)b * T * C + c;
    for (int t = T - 1; t >= 0; --t) {
        size_t off = base + (size_t)t * C;
        float kt  = k[off],  vt  = v[off];
        float afo = af[off], bfo = bfv[off], pfo = pf[off];
        float pc = u + kt;
        float m  = fmaxf(fmaxf(pfo, p), pc);
        float ef = __expf(pfo - m);
        float eb = __expf(p   - m);
        float ec = __expf(pc  - m);
        float num = afo * ef + a  * eb + vt * ec;
        float den = bfo * ef + bb * eb + ec;
        outb[off] = f2bf(sr[off] * (num / den));
        // advance suffix state to include token t
        float m2 = fmaxf(p - w, kt);
        float e1 = __expf(p - w - m2);
        float e2 = __expf(kt - m2);
        a  = a * e1 + vt * e2;
        bb = bb * e1 + e2;
        p  = m2;
    }
}

// ---------------------------------------------------------------------------
// launch
// ---------------------------------------------------------------------------
extern "C" void kernel_launch(void* const* d_in, const int* in_sizes, int n_in,
                              void* d_out, int out_size, void* d_ws, size_t ws_size,
                              hipStream_t stream) {
    const float* x     = (const float*)d_in[0];
    const float* decay = (const float*)d_in[1];
    const float* first = (const float*)d_in[2];
    const float* mk    = (const float*)d_in[3];
    const float* mv    = (const float*)d_in[4];
    const float* mr    = (const float*)d_in[5];
    const float* Wk    = (const float*)d_in[6];
    const float* Wv    = (const float*)d_in[7];
    const float* Wr    = (const float*)d_in[8];
    const float* Wo    = (const float*)d_in[9];

    const int C = in_sizes[1];                 // 768
    const int T = 4096;                        // patch_resolution^2 (64*64)
    const long long N = (long long)in_sizes[0];
    const int B = (int)(N / ((long long)T * C));
    const int M = B * T;                       // GEMM rows

    // ---- carve workspace (256B aligned sections)
    uint8_t* p = (uint8_t*)d_ws;
    auto carve = [&](size_t bytes) -> void* {
        void* r = (void*)p;
        p += (bytes + 255) & ~(size_t)255;
        return r;
    };
    unsigned short* xk  = (unsigned short*)carve(2 * (size_t)N);
    unsigned short* xv  = (unsigned short*)carve(2 * (size_t)N);
    unsigned short* xr  = (unsigned short*)carve(2 * (size_t)N);
    unsigned short* wkb = (unsigned short*)carve(2 * (size_t)C * C);
    unsigned short* wvb = (unsigned short*)carve(2 * (size_t)C * C);
    unsigned short* wrb = (unsigned short*)carve(2 * (size_t)C * C);
    unsigned short* wob = (unsigned short*)carve(2 * (size_t)C * C);
    float* kf  = (float*)carve(4 * (size_t)N);
    float* vf  = (float*)carve(4 * (size_t)N);
    float* srf = (float*)carve(4 * (size_t)N);
    float* afp = (float*)carve(4 * (size_t)N);
    float* bfp = (float*)carve(4 * (size_t)N);
    float* pfp = (float*)carve(4 * (size_t)N);
    unsigned short* srr = (unsigned short*)carve(2 * (size_t)N);

    // 1) weights -> bf16
    {
        int n = C * C;
        int total = 4 * n;
        convert_weights_kernel<<<(total + 255) / 256, 256, 0, stream>>>(
            Wk, Wv, Wr, Wo, wkb, wvb, wrb, wob, n);
    }
    // 2) token shift + mix -> bf16 activations
    {
        long long blocks = (N + 255) / 256;
        mix_shift_kernel<<<(unsigned)blocks, 256, 0, stream>>>(
            x, mk, mv, mr, xk, xv, xr, T, C, N);
    }
    // 3) k / v / sigmoid(r) GEMMs
    {
        dim3 grid(M / 128, C / 128);
        gemm_bf16_wmma_kernel<<<grid, 256, 0, stream>>>(xk, wkb, kf,  M, C, C, 0);
        gemm_bf16_wmma_kernel<<<grid, 256, 0, stream>>>(xv, wvb, vf,  M, C, C, 0);
        gemm_bf16_wmma_kernel<<<grid, 256, 0, stream>>>(xr, wrb, srf, M, C, C, 1);
    }
    // 4) bidirectional WKV scans
    {
        int lanes = B * C;
        scan_fwd_kernel<<<(lanes + 255) / 256, 256, 0, stream>>>(
            kf, vf, decay, afp, bfp, pfp, B, T, C);
        scan_bwd_combine_kernel<<<(lanes + 255) / 256, 256, 0, stream>>>(
            kf, vf, decay, first, afp, bfp, pfp, srf, srr, B, T, C);
    }
    // 5) output projection -> f32 d_out
    {
        dim3 grid(M / 128, C / 128);
        gemm_bf16_wmma_kernel<<<grid, 256, 0, stream>>>(srr, wob, (float*)d_out, M, C, C, 0);
    }
}
